// DK_spatial_attention_v2_80951543595370
// MI455X (gfx1250) — compile-verified, matrix operand order checked by reading, not run on hardware
//
#include <hip/hip_runtime.h>
#include <math.h>

// ---------------------------------------------------------------------------
// Types for CDNA5 WMMA (wave32): bf16 A/B fragments, f32 accumulator
// ---------------------------------------------------------------------------
typedef __attribute__((ext_vector_type(16))) __bf16 v16bf;
typedef __attribute__((ext_vector_type(8)))  float  v8f;

// Pack two f32 -> two bf16 in one dword (lowers to v_cvt_pk_bf16_f32)
__device__ __forceinline__ unsigned int pk_bf16(float a, float b) {
    union { __bf16 h[2]; unsigned int u; } t;
    t.h[0] = (__bf16)a; t.h[1] = (__bf16)b;
    return t.u;
}

// Fragment-order LDS swizzle (ISA 7.12.2, 16-bit 16x32 A / 32x16 B):
//   element (row m or col n, k) lives at lane = (x&15) + ((k>>3)&1 ? 16 : 0),
//   position e = (k&7) + (k>=16 ? 8 : 0) within that lane's 16 bf16 values.
// A quad of 4 consecutive k (k = 4q) shares one lane and 4 consecutive e:
//   lane_hi = (q>>1)&1 ; e0/4 -> u64 slot = (q&1) + (q>=4 ? 2 : 0)

// ---------------------------------------------------------------------------
// conv1: 3x3 stride-2 pad-1, 64 -> 64, in 8x64x256x256, out 8x64x128x128
// Implicit GEMM: one block = one output row (b,ho): M=128 (wo), N=64, K=576
// 8 waves; wave w owns output rows [16w,16w+16); K chunked 9 taps x 2 halves
// ---------------------------------------------------------------------------
__global__ __launch_bounds__(256)
void conv1_s2_wmma(const float* __restrict__ X, const float* __restrict__ W,
                   const float* __restrict__ bias, float* __restrict__ Y) {
    // fragment-order LDS, double buffered: A = 8 waves x 32 lanes x 4 u64,
    // B = 4 ntiles x 32 lanes x 4 u64
    __shared__ __align__(32) unsigned long long Af[2][8 * 32 * 4];
    __shared__ __align__(32) unsigned long long Bf[2][4 * 32 * 4];
    const int tid  = threadIdx.x;
    const int b    = blockIdx.x >> 7;
    const int ho   = blockIdx.x & 127;
    const int lane = tid & 31, wave = tid >> 5;

    v8f acc[4];
    #pragma unroll
    for (int nt = 0; nt < 4; ++nt)
        acc[nt] = (v8f){0.f, 0.f, 0.f, 0.f, 0.f, 0.f, 0.f, 0.f};

    for (int chunk = 0; chunk < 18; ++chunk) {
        const int tap  = chunk >> 1;          // 0..8
        const int half = chunk & 1;           // channel half
        const int buf  = chunk & 1;           // LDS double buffer
        const int kh   = tap / 3, kw = tap % 3;
        const int hin  = 2 * ho + kh - 1;
        const int hc   = hin < 0 ? 0 : (hin > 255 ? 255 : hin);
        const float vh = ((unsigned)hin < 256u) ? 1.f : 0.f;
        const int cbase = half * 32;
        {   // speculative prefetch of next input row
            int hpf = hc + 1; if (hpf > 255) hpf = 255;
            __builtin_prefetch(X + (((size_t)b * 64 + cbase) * 256 + hpf) * 256, 0, 1);
        }
        // ---- stage A: 128 rows x 32 chans as 1024 channel-quads, 4/thread
        #pragma unroll
        for (int i = 0; i < 4; ++i) {
            int idx = tid + 256 * i;
            int m   = idx & 127;              // output col (wo)
            int q   = idx >> 7;               // channel quad 0..7
            int win = 2 * m + kw - 1;
            int wc  = win < 0 ? 0 : (win > 255 ? 255 : win);
            float vw = ((unsigned)win < 256u) ? vh : 0.f;
            const float* xp = X + (((size_t)b * 64 + cbase + 4 * q) * 256 + hc) * 256 + wc;
            float f0 = xp[0]      * vw;
            float f1 = xp[65536]  * vw;
            float f2 = xp[131072] * vw;
            float f3 = xp[196608] * vw;
            unsigned long long pk = ((unsigned long long)pk_bf16(f2, f3) << 32)
                                  |  (unsigned long long)pk_bf16(f0, f1);
            int fl   = (m & 15) + (((q >> 1) & 1) ? 16 : 0);
            int slot = (q & 1) + ((q >= 4) ? 2 : 0);
            Af[buf][((m >> 4) * 32 + fl) * 4 + slot] = pk;
        }
        // ---- stage B: 32 chans x 64 couts as 512 quads, 2/thread
        #pragma unroll
        for (int i = 0; i < 2; ++i) {
            int idx = tid + 256 * i;
            int n   = idx & 63;
            int q   = idx >> 6;               // 0..7
            const float* wp = W + ((size_t)n * 64 + cbase + 4 * q) * 9 + tap;
            float f0 = wp[0], f1 = wp[9], f2 = wp[18], f3 = wp[27];
            unsigned long long pk = ((unsigned long long)pk_bf16(f2, f3) << 32)
                                  |  (unsigned long long)pk_bf16(f0, f1);
            int fl   = (n & 15) + (((q >> 1) & 1) ? 16 : 0);
            int slot = (q & 1) + ((q >= 4) ? 2 : 0);
            Bf[buf][((n >> 4) * 32 + fl) * 4 + slot] = pk;
        }
        __syncthreads();
        // ---- fragments: one contiguous 32B read per lane -> 2x ds_load_b128
        v16bf a = *(const v16bf*)&Af[buf][(wave * 32 + lane) * 4];
        #pragma unroll
        for (int nt = 0; nt < 4; ++nt) {
            v16bf bm = *(const v16bf*)&Bf[buf][(nt * 32 + lane) * 4];
            acc[nt] = __builtin_amdgcn_wmma_f32_16x16x32_bf16(
                false, a, false, bm, (short)0, acc[nt], false, false);
        }
    }
    // ---- epilogue: bias + ReLU (C/D layout: lane<16 M=r, lane>=16 M=r+8)
    #pragma unroll
    for (int nt = 0; nt < 4; ++nt) {
        int cout = nt * 16 + (lane & 15);
        float bv = bias[cout];
        #pragma unroll
        for (int r = 0; r < 8; ++r) {
            int wo = wave * 16 + ((lane < 16) ? r : (r + 8));
            float v = acc[nt][r] + bv;
            v = v > 0.f ? v : 0.f;
            Y[(((size_t)b * 64 + cout) * 128 + ho) * 128 + wo] = v;
        }
    }
}

// ---------------------------------------------------------------------------
// conv2: 3x3 stride-1 pad-1, 64 -> 256, in 8x64x128x128; store fused with
// pixel_shuffle(r=2) into 8x64x256x256
// ---------------------------------------------------------------------------
__global__ __launch_bounds__(256)
void conv2_wmma_shuffle(const float* __restrict__ X, const float* __restrict__ W,
                        const float* __restrict__ bias, float* __restrict__ Y) {
    __shared__ __align__(32) unsigned long long Af[2][8 * 32 * 4];
    __shared__ __align__(32) unsigned long long Bf[2][4 * 32 * 4];
    const int tid   = threadIdx.x;
    const int b     = blockIdx.x >> 7;
    const int ho    = blockIdx.x & 127;
    const int nbase = blockIdx.y * 64;
    const int lane  = tid & 31, wave = tid >> 5;

    v8f acc[4];
    #pragma unroll
    for (int nt = 0; nt < 4; ++nt)
        acc[nt] = (v8f){0.f, 0.f, 0.f, 0.f, 0.f, 0.f, 0.f, 0.f};

    for (int chunk = 0; chunk < 18; ++chunk) {
        const int tap  = chunk >> 1;
        const int half = chunk & 1;
        const int buf  = chunk & 1;
        const int kh   = tap / 3, kw = tap % 3;
        const int hin  = ho + kh - 1;
        const int hc   = hin < 0 ? 0 : (hin > 127 ? 127 : hin);
        const float vh = ((unsigned)hin < 128u) ? 1.f : 0.f;
        const int cbase = half * 32;
        {
            int hpf = hc + 1; if (hpf > 127) hpf = 127;
            __builtin_prefetch(X + (((size_t)b * 64 + cbase) * 128 + hpf) * 128, 0, 1);
        }
        #pragma unroll
        for (int i = 0; i < 4; ++i) {
            int idx = tid + 256 * i;
            int m   = idx & 127;
            int q   = idx >> 7;
            int win = m + kw - 1;
            int wc  = win < 0 ? 0 : (win > 127 ? 127 : win);
            float vw = ((unsigned)win < 128u) ? vh : 0.f;
            const float* xp = X + (((size_t)b * 64 + cbase + 4 * q) * 128 + hc) * 128 + wc;
            float f0 = xp[0]     * vw;
            float f1 = xp[16384] * vw;
            float f2 = xp[32768] * vw;
            float f3 = xp[49152] * vw;
            unsigned long long pk = ((unsigned long long)pk_bf16(f2, f3) << 32)
                                  |  (unsigned long long)pk_bf16(f0, f1);
            int fl   = (m & 15) + (((q >> 1) & 1) ? 16 : 0);
            int slot = (q & 1) + ((q >= 4) ? 2 : 0);
            Af[buf][((m >> 4) * 32 + fl) * 4 + slot] = pk;
        }
        #pragma unroll
        for (int i = 0; i < 2; ++i) {
            int idx = tid + 256 * i;
            int n   = idx & 63;
            int q   = idx >> 6;
            const float* wp = W + ((size_t)(nbase + n) * 64 + cbase + 4 * q) * 9 + tap;
            float f0 = wp[0], f1 = wp[9], f2 = wp[18], f3 = wp[27];
            unsigned long long pk = ((unsigned long long)pk_bf16(f2, f3) << 32)
                                  |  (unsigned long long)pk_bf16(f0, f1);
            int fl   = (n & 15) + (((q >> 1) & 1) ? 16 : 0);
            int slot = (q & 1) + ((q >= 4) ? 2 : 0);
            Bf[buf][((n >> 4) * 32 + fl) * 4 + slot] = pk;
        }
        __syncthreads();
        v16bf a = *(const v16bf*)&Af[buf][(wave * 32 + lane) * 4];
        #pragma unroll
        for (int nt = 0; nt < 4; ++nt) {
            v16bf bm = *(const v16bf*)&Bf[buf][(nt * 32 + lane) * 4];
            acc[nt] = __builtin_amdgcn_wmma_f32_16x16x32_bf16(
                false, a, false, bm, (short)0, acc[nt], false, false);
        }
    }
    // epilogue: bias add, pixel-shuffle store (cout = 4*c + 2*ry + rx)
    #pragma unroll
    for (int nt = 0; nt < 4; ++nt) {
        int cout = nbase + nt * 16 + (lane & 15);
        int c  = cout >> 2;
        int ry = (cout >> 1) & 1;
        int rx = cout & 1;
        float bv = bias[cout];
        #pragma unroll
        for (int r = 0; r < 8; ++r) {
            int wo = wave * 16 + ((lane < 16) ? r : (r + 8));
            float v = acc[nt][r] + bv;
            Y[(((size_t)b * 64 + c) * 256 + (2 * ho + ry)) * 256 + (2 * wo + rx)] = v;
        }
    }
}

// ---------------------------------------------------------------------------
// Global average pool: h (8,64,128,128) -> feat (8,64)
// ---------------------------------------------------------------------------
__global__ __launch_bounds__(256)
void gap_kernel(const float* __restrict__ H, float* __restrict__ feat) {
    __shared__ float red[256];
    const int bc  = blockIdx.x;
    const int tid = threadIdx.x;
    const float* plane = H + (size_t)bc * 16384;
    float s = 0.f;
    #pragma unroll 8
    for (int i = 0; i < 64; ++i) s += plane[tid + 256 * i];
    red[tid] = s;
    __syncthreads();
    for (int off = 128; off > 0; off >>= 1) {
        if (tid < off) red[tid] += red[tid + off];
        __syncthreads();
    }
    if (tid == 0) feat[bc] = red[0] * (1.f / 16384.f);
}

// ---------------------------------------------------------------------------
// FC offsets + bilinear kernel sampling (tiny; one block)
// ---------------------------------------------------------------------------
__global__ __launch_bounds__(256)
void dkc_off_samp(const float* __restrict__ feat, const float* __restrict__ fcw,
                  const float* __restrict__ fcb, const float* __restrict__ w4,
                  float* __restrict__ off, float* __restrict__ samp) {
    const int tid = threadIdx.x;
    if (tid < 144) {
        int b = tid / 18, j = tid % 18;
        float s = fcb[j];
        for (int c = 0; c < 64; ++c) s += feat[b * 64 + c] * fcw[j * 64 + c];
        off[b * 18 + j] = s;
    }
    __syncthreads();
    for (int idx = tid; idx < 8 * 64 * 9; idx += 256) {
        int b   = idx / 576;
        int rem = idx % 576;
        int c   = rem / 9;
        int p   = rem % 9;
        int i   = p / 3, j = p % 3;
        float cy = (float)i + 0.5f + off[b * 18 + p];
        float cx = (float)j + 0.5f + off[b * 18 + 9 + p];
        float y0f = floorf(cy), x0f = floorf(cx);
        float wy = cy - y0f, wx = cx - x0f;
        int y0 = (int)y0f, x0 = (int)x0f;
        const float* w2 = w4 + c * 16;
        auto g = [&](int iy, int ix) -> float {
            if (iy < 0 || iy > 3 || ix < 0 || ix > 3) return 0.f;
            return w2[iy * 4 + ix];
        };
        samp[idx] = g(y0, x0) * (1.f - wy) * (1.f - wx)
                  + g(y0, x0 + 1) * (1.f - wy) * wx
                  + g(y0 + 1, x0) * wy * (1.f - wx)
                  + g(y0 + 1, x0 + 1) * wy * wx;
    }
}

// ---------------------------------------------------------------------------
// Depthwise 3x3 pad-1 with per-(b,c) dynamic kernels + bias + ReLU
// ---------------------------------------------------------------------------
__global__ __launch_bounds__(256)
void dkc_depthwise(const float* __restrict__ H, const float* __restrict__ samp,
                   const float* __restrict__ bias, float* __restrict__ O) {
    const int bc  = blockIdx.x;
    const int c   = bc & 63;
    const int tid = threadIdx.x;
    const int pix = blockIdx.y * 256 + tid;
    const int y   = pix >> 7, xw = pix & 127;
    const float* plane = H + (size_t)bc * 16384;
    float k[9];
    #pragma unroll
    for (int i = 0; i < 9; ++i) k[i] = samp[bc * 9 + i];
    float s = bias[c];
    #pragma unroll
    for (int dy = -1; dy <= 1; ++dy) {
        #pragma unroll
        for (int dx = -1; dx <= 1; ++dx) {
            int yy = y + dy, xx = xw + dx;
            float v = 0.f;
            if ((unsigned)yy < 128u && (unsigned)xx < 128u) v = plane[yy * 128 + xx];
            s += v * k[(dy + 1) * 3 + (dx + 1)];
        }
    }
    O[(size_t)bc * 16384 + pix] = s > 0.f ? s : 0.f;
}

// ---------------------------------------------------------------------------
// Final: att = sigmoid(1x1 conv over shuffled 64ch) ; out = x * att
// ---------------------------------------------------------------------------
__global__ __launch_bounds__(256)
void final_att(const float* __restrict__ X, const float* __restrict__ HS,
               const float* __restrict__ w3, const float* __restrict__ b3,
               float* __restrict__ OUT) {
    const int idx = blockIdx.x * 256 + threadIdx.x;
    const int b   = idx >> 16;
    const int yx  = idx & 65535;
    const float* hp = HS + (size_t)b * 64 * 65536 + yx;
    float att = b3[0];
    #pragma unroll 8
    for (int c = 0; c < 64; ++c) att += hp[(size_t)c * 65536] * w3[c];
    att = 1.f / (1.f + expf(-att));
    const float* xp = X + (size_t)b * 64 * 65536 + yx;
    float* op = OUT + (size_t)b * 64 * 65536 + yx;
    #pragma unroll 8
    for (int c = 0; c < 64; ++c) op[(size_t)c * 65536] = xp[(size_t)c * 65536] * att;
}

// ---------------------------------------------------------------------------
// Host launcher
// ---------------------------------------------------------------------------
extern "C" void kernel_launch(void* const* d_in, const int* in_sizes, int n_in,
                              void* d_out, int out_size, void* d_ws, size_t ws_size,
                              hipStream_t stream) {
    (void)in_sizes; (void)n_in; (void)out_size; (void)ws_size;
    const float* x    = (const float*)d_in[0];
    const float* c1w  = (const float*)d_in[1];
    const float* c1b  = (const float*)d_in[2];
    const float* dkcw = (const float*)d_in[3];
    const float* dkcb = (const float*)d_in[4];
    const float* fcw  = (const float*)d_in[5];
    const float* fcb  = (const float*)d_in[6];
    const float* c2w  = (const float*)d_in[7];
    const float* c2b  = (const float*)d_in[8];
    const float* c3w  = (const float*)d_in[9];
    const float* c3b  = (const float*)d_in[10];
    float* out = (float*)d_out;

    char* ws = (char*)d_ws;
    float* hA   = (float*)(ws);
    float* hB   = (float*)(ws + (size_t)33554432);
    float* hs   = (float*)(ws + (size_t)67108864);
    float* feat = (float*)(ws + (size_t)201326592);
    float* off  = (float*)(ws + (size_t)201326592 + 4096);
    float* samp = (float*)(ws + (size_t)201326592 + 8192);

    conv1_s2_wmma<<<dim3(1024), dim3(256), 0, stream>>>(x, c1w, c1b, hA);

    float* cur = hA; float* nxt = hB;
    for (int i = 0; i < 5; ++i) {
        gap_kernel<<<dim3(512), dim3(256), 0, stream>>>(cur, feat);
        dkc_off_samp<<<dim3(1), dim3(256), 0, stream>>>(
            feat, fcw + (size_t)i * 18 * 64, fcb + (size_t)i * 18,
            dkcw + (size_t)i * 64 * 16, off, samp);
        dkc_depthwise<<<dim3(512, 64), dim3(256), 0, stream>>>(
            cur, samp, dkcb + (size_t)i * 64, nxt);
        float* t = cur; cur = nxt; nxt = t;
    }

    conv2_wmma_shuffle<<<dim3(1024, 4), dim3(256), 0, stream>>>(cur, c2w, c2b, hs);
    final_att<<<dim3(2048), dim3(256), 0, stream>>>(x, hs, c3w, c3b, out);
}
